// CustomPatchEmbedding_79852031967428
// MI455X (gfx1250) — compile-verified, matrix operand order checked by reading, not run on hardware
//
#include <hip/hip_runtime.h>

typedef __attribute__((ext_vector_type(16))) __bf16 v16bf;
typedef __attribute__((ext_vector_type(8)))  float  v8f;

#define DIM    256     // embed dim
#define NTOT   288     // NF + NC rows per image in output
#define IMG_H  512
#define IMG_W  512
#define KC     768     // K-chunk staged in LDS (whole K for fine, 16 chunks for coarse)
#define KPAD   8       // +16B per row -> conflict-free ds_load_b128 across 16 rows

// Load a 16-element bf16 WMMA fragment: two contiguous 16B runs at +0 and +16
// (per the CDNA5 16-bit A/B VGPR layout, a lane's K values are {kb..kb+7, kb+16..kb+23}).
static __device__ inline v16bf load_frag16(const __bf16* p) {
  union { int4 q[2]; v16bf v; } u;
  u.q[0] = *reinterpret_cast<const int4*>(p);
  u.q[1] = *reinterpret_cast<const int4*>(p + 16);
  return u.v;
}

// fp32 -> bf16 hi/lo split of a weight matrix (row-major, D x K)
__global__ void split_bf16_kernel(const float* __restrict__ w,
                                  __bf16* __restrict__ hi,
                                  __bf16* __restrict__ lo, int n) {
  int i = blockIdx.x * blockDim.x + threadIdx.x;
  if (i < n) {
    float v = w[i];
    __bf16 h = (__bf16)v;            // RNE
    hi[i] = h;
    lo[i] = (__bf16)(v - (float)h);  // residual
  }
}

// One block: 16 output rows (patches) x (WAVES*16) output cols.
// LOG2P: log2(patch size). NPATCH: patches/image. NOFF: row offset in output.
// KTOT: C*P*P. WAVES: waves per block (each wave owns a 16-col slice of D).
template<int LOG2P, int NPATCH, int NOFF, int KTOT, int WAVES>
__global__ __launch_bounds__(WAVES * 32)
void patch_embed_wmma(const float* __restrict__ images,
                      const __bf16* __restrict__ Whi,
                      const __bf16* __restrict__ Wlo,
                      const float* __restrict__ bias,
                      const int*   __restrict__ xy,
                      float*       __restrict__ out) {
  constexpr int P   = 1 << LOG2P;
  constexpr int TPI = NPATCH / 16;              // row tiles per image

  __shared__ __bf16 sAhi[16 * (KC + KPAD)];
  __shared__ __bf16 sAlo[16 * (KC + KPAD)];
  __shared__ int sx[16], sy[16];

  const int tid   = threadIdx.x;
  const int tile  = blockIdx.x;
  const int b     = tile / TPI;
  const int pbase = (tile % TPI) * 16;
  const int lane  = tid & 31;
  const int m     = lane & 15;                  // A row / B col within tile
  const int khalf = (lane >> 4) * 8;            // lane-half K offset
  const int dBase = (blockIdx.y * WAVES + (tid >> 5)) * 16;

  if (tid < 16) {
    sx[tid] = xy[(b * NPATCH + pbase + tid) * 2 + 0];  // x
    sy[tid] = xy[(b * NPATCH + pbase + tid) * 2 + 1];  // y
  }
  __syncthreads();

  v8f acc = {};
  const __bf16* arow_hi = &sAhi[m * (KC + KPAD)];
  const __bf16* arow_lo = &sAlo[m * (KC + KPAD)];

  for (int kc0 = 0; kc0 < KTOT; kc0 += KC) {
    // Cooperative gather: 16 patches x KC elems from image -> LDS (bf16 hi/lo).
    for (int f = tid; f < 16 * KC; f += WAVES * 32) {
      int row = f / KC;
      int kk  = f - row * KC;
      int k   = kc0 + kk;                        // k = c*P*P + py*P + px
      int c   = k >> (2 * LOG2P);
      int py  = (k >> LOG2P) & (P - 1);
      int px  = k & (P - 1);
      float v = images[(((size_t)(b * 3 + c) * IMG_H) + sy[row] + py) * IMG_W
                       + sx[row] + px];
      __bf16 h = (__bf16)v;
      sAhi[row * (KC + KPAD) + kk] = h;
      sAlo[row * (KC + KPAD) + kk] = (__bf16)(v - (float)h);
    }
    __syncthreads();

    const __bf16* wrow_hi = Whi + (size_t)(dBase + m) * KTOT + kc0 + khalf;
    const __bf16* wrow_lo = Wlo + (size_t)(dBase + m) * KTOT + kc0 + khalf;

    #pragma unroll 2
    for (int k0 = 0; k0 < KC; k0 += 32) {
      v16bf ah = load_frag16(arow_hi + k0 + khalf);
      v16bf al = load_frag16(arow_lo + k0 + khalf);
      v16bf bh = load_frag16(wrow_hi + k0);
      v16bf bl = load_frag16(wrow_lo + k0);
      // (ahi+alo)*(bhi+blo), all accumulated in f32
      acc = __builtin_amdgcn_wmma_f32_16x16x32_bf16(false, ah, false, bh, (short)0, acc, false, false);
      acc = __builtin_amdgcn_wmma_f32_16x16x32_bf16(false, ah, false, bl, (short)0, acc, false, false);
      acc = __builtin_amdgcn_wmma_f32_16x16x32_bf16(false, al, false, bh, (short)0, acc, false, false);
      acc = __builtin_amdgcn_wmma_f32_16x16x32_bf16(false, al, false, bl, (short)0, acc, false, false);
    }
    __syncthreads();
  }

  // C/D layout: VGPR r holds M=r (lanes 0-15) or M=r+8 (lanes 16-31); N = lane&15.
  float bv = bias[dBase + m];
  int mofs = (lane >> 4) * 8;
  #pragma unroll
  for (int r = 0; r < 8; ++r) {
    int n = NOFF + pbase + r + mofs;
    out[((size_t)b * NTOT + n) * DIM + dBase + m] = acc[r] + bv;
  }
}

extern "C" void kernel_launch(void* const* d_in, const int* in_sizes, int n_in,
                              void* d_out, int out_size, void* d_ws, size_t ws_size,
                              hipStream_t stream) {
  // setup_inputs order: images, W_fine, b_fine, W_coarse, b_coarse, fine_xy, coarse_xy
  const float* images   = (const float*)d_in[0];
  const float* W_fine   = (const float*)d_in[1];
  const float* b_fine   = (const float*)d_in[2];
  const float* W_coarse = (const float*)d_in[3];
  const float* b_coarse = (const float*)d_in[4];
  const int*   fine_xy   = (const int*)d_in[5];
  const int*   coarse_xy = (const int*)d_in[6];
  float* out = (float*)d_out;

  const int KF = 3 * 16 * 16;    // 768
  const int KCo = 3 * 64 * 64;   // 12288
  const int nWf = DIM * KF;      // 196608
  const int nWc = DIM * KCo;     // 3145728

  // workspace: [Wf_hi | Wf_lo | Wc_hi | Wc_lo] as bf16 (~12.8 MB)
  __bf16* Wf_hi = (__bf16*)d_ws;
  __bf16* Wf_lo = Wf_hi + nWf;
  __bf16* Wc_hi = Wf_lo + nWf;
  __bf16* Wc_lo = Wc_hi + nWc;

  split_bf16_kernel<<<(nWf + 255) / 256, 256, 0, stream>>>(W_fine, Wf_hi, Wf_lo, nWf);
  split_bf16_kernel<<<(nWc + 255) / 256, 256, 0, stream>>>(W_coarse, Wc_hi, Wc_lo, nWc);

  // Fine: 32*256/16 = 512 row tiles, 16 waves cover all 256 cols.
  patch_embed_wmma<4, 256, 0, 768, 16>
      <<<dim3(512, 1), 512, 0, stream>>>(images, Wf_hi, Wf_lo, b_fine, fine_xy, out);

  // Coarse: 32*32/16 = 64 row tiles; split D in 2 for more blocks (128 total).
  patch_embed_wmma<6, 32, 256, 12288, 8>
      <<<dim3(64, 2), 256, 0, stream>>>(images, Wc_hi, Wc_lo, b_coarse, coarse_xy, out);
}
